// TestModel_11072425689833
// MI455X (gfx1250) — 8-bit matrix operand order repaired and checked by reading, compiled, not run on hardware
//
#include <hip/hip_runtime.h>

#define Hdim 4096
#define Tdim 4096

typedef __attribute__((ext_vector_type(16))) int          v16i;
typedef __attribute__((ext_vector_type(8)))  float        v8f;
typedef __attribute__((ext_vector_type(4)))  unsigned int v4u;
typedef __attribute__((ext_vector_type(8)))  int          v8i;
typedef __attribute__((ext_vector_type(4)))  int          v4i;

#if defined(__has_builtin)
#if __has_builtin(__builtin_amdgcn_tensor_load_to_lds)
#define HAVE_TDM 1
#endif
#endif

__device__ __forceinline__ unsigned char f32_to_e4m3(float f) {
    unsigned int x = __float_as_uint(f);
    unsigned int sign = (x >> 31) << 7;
    x &= 0x7fffffffu;
    float a = __uint_as_float(x);
    unsigned int code;
    if (a < 0.015625f) {
        int m = (int)rintf(a * 512.0f);
        code = (unsigned int)m;
    } else {
        unsigned int lsb = (x >> 20) & 1u;
        unsigned int r = (x + 0x7ffffu + lsb) >> 20;
        int e = (int)(r >> 3) - 127;
        code = (e > 8) ? 0x7eu : (((unsigned int)(e + 7) << 3) | (r & 7u));
    }
    return (unsigned char)(sign | code);
}

__device__ __forceinline__ unsigned char quant_e4m3(float y, float inv_scale) {
    float z = y * inv_scale;
    z = fminf(fmaxf(z, -448.0f), 448.0f);
    return f32_to_e4m3(z);
}

__global__ __launch_bounds__(256)
void relu_rms_quant_kernel(const float* __restrict__ x,
                           const float* __restrict__ nw,
                           const float* __restrict__ scale,
                           float* __restrict__ resid,
                           unsigned char* __restrict__ qout) {
    const int row = blockIdx.x;
    const int tid = threadIdx.x;
    const float* xr = x + (size_t)row * Hdim;
    float v[16];
    float ss = 0.f;
#pragma unroll
    for (int i = 0; i < 16; ++i) {
        float t = fmaxf(xr[tid + i * 256], 0.f);
        v[i] = t;
        ss += t * t;
    }
    __shared__ float red[256];
    red[tid] = ss;
    __syncthreads();
    for (int s2 = 128; s2 > 0; s2 >>= 1) {
        if (tid < s2) red[tid] += red[tid + s2];
        __syncthreads();
    }
    float rinv = rsqrtf(red[0] * (1.0f / Hdim) + 1e-6f);
    float inv_scale = 1.0f / scale[0];
    float* rr = resid + (size_t)row * Hdim;
    unsigned char* qr = qout + (size_t)row * Hdim;
#pragma unroll
    for (int i = 0; i < 16; ++i) {
        int idx = tid + i * 256;
        rr[idx] = v[i];
        qr[idx] = quant_e4m3(v[i] * rinv * nw[idx], inv_scale);
    }
}

__global__ __launch_bounds__(256)
void rms_quant_kernel(const float* __restrict__ resid,
                      const float* __restrict__ nw,
                      const float* __restrict__ scale,
                      unsigned char* __restrict__ qout) {
    const int row = blockIdx.x;
    const int tid = threadIdx.x;
    const float* xr = resid + (size_t)row * Hdim;
    float v[16];
    float ss = 0.f;
#pragma unroll
    for (int i = 0; i < 16; ++i) {
        float t = xr[tid + i * 256];
        v[i] = t;
        ss += t * t;
    }
    __shared__ float red[256];
    red[tid] = ss;
    __syncthreads();
    for (int s2 = 128; s2 > 0; s2 >>= 1) {
        if (tid < s2) red[tid] += red[tid + s2];
        __syncthreads();
    }
    float rinv = rsqrtf(red[0] * (1.0f / Hdim) + 1e-6f);
    float inv_scale = 1.0f / scale[0];
    unsigned char* qr = qout + (size_t)row * Hdim;
#pragma unroll
    for (int i = 0; i < 16; ++i) {
        int idx = tid + i * 256;
        qr[idx] = quant_e4m3(v[i] * rinv * nw[idx], inv_scale);
    }
}

__global__ __launch_bounds__(256)
void rms_out_kernel(const float* __restrict__ resid,
                    const float* __restrict__ nw,
                    float* __restrict__ out) {
    const int row = blockIdx.x;
    const int tid = threadIdx.x;
    const float* xr = resid + (size_t)row * Hdim;
    float v[16];
    float ss = 0.f;
#pragma unroll
    for (int i = 0; i < 16; ++i) {
        float t = xr[tid + i * 256];
        v[i] = t;
        ss += t * t;
    }
    __shared__ float red[256];
    red[tid] = ss;
    __syncthreads();
    for (int s2 = 128; s2 > 0; s2 >>= 1) {
        if (tid < s2) red[tid] += red[tid + s2];
        __syncthreads();
    }
    float rinv = rsqrtf(red[0] * (1.0f / Hdim) + 1e-6f);
    float* orow = out + (size_t)row * Hdim;
#pragma unroll
    for (int i = 0; i < 16; ++i) {
        int idx = tid + i * 256;
        orow[idx] = v[i] * rinv * nw[idx];
    }
}

__global__ __launch_bounds__(256)
void wconv_kernel(const float* __restrict__ W, unsigned char* __restrict__ wq) {
    __shared__ unsigned char tile[32][33];
    const int nb = blockIdx.x * 32;
    const int kb = blockIdx.y * 32;
    const int tx = threadIdx.x, ty = threadIdx.y;
    for (int r = ty; r < 32; r += 8) {
        float f = W[(size_t)(kb + r) * Hdim + nb + tx];
        tile[r][tx] = f32_to_e4m3(fminf(fmaxf(f, -448.f), 448.f));
    }
    __syncthreads();
    for (int r = ty; r < 32; r += 8) {
        wq[(size_t)(nb + r) * Hdim + kb + tx] = tile[tx][r];
    }
}

#define KSTEP   128
#define PITCH   144
#define AROWS   128
#define BROWS   64
#define ASZ     (AROWS * PITCH)
#define BSZ     (BROWS * PITCH)
#define BUFSZ   (ASZ + BSZ)

#ifdef HAVE_TDM
__device__ __forceinline__ void tdm_load_tile(const unsigned char* g, unsigned lds_off, int rows) {
    unsigned long long ga = (unsigned long long)(size_t)g;
    v4u g0 = { 1u, lds_off, (unsigned)(ga & 0xffffffffu),
               (unsigned)(((ga >> 32) & 0x01ffffffu) | 0x80000000u) };
    v8i g1 = { 0x07100000, 0x10000000, 0x10000000, 0x00800000, rows, 0x1000, 0, 0 };
    v4i z4 = { 0, 0, 0, 0 };
#if __has_include(<hip/amd_detail/amd_gfx1250_TDM.h>)
    v8i z8 = { 0, 0, 0, 0, 0, 0, 0, 0 };
    __builtin_amdgcn_tensor_load_to_lds(g0, g1, z4, z4, z8, 0);
#else
    __builtin_amdgcn_tensor_load_to_lds(g0, g1, z4, z4, 0);
#endif
}
#endif

__global__ __launch_bounds__(256)
void gemm_fp8_kernel(const unsigned char* __restrict__ q,
                     const unsigned char* __restrict__ wt,
                     float* __restrict__ resid,
                     const float* __restrict__ scale,
                     const float* __restrict__ wscale,
                     int l) {
    __shared__ unsigned char smem[2 * BUFSZ];
    const int tid  = threadIdx.x;
    const int lane = tid & 31;
    const int wv   = __builtin_amdgcn_readfirstlane((int)tid) >> 5;
    const int l15  = lane & 15;
    const int lhi  = lane >> 4;
    const int wm   = wv & 3;
    const int wn   = wv >> 2;
    const int mblk = blockIdx.y * 128;
    const int nblk = blockIdx.x * 64;

    v8f acc[2][2];
#pragma unroll
    for (int mi = 0; mi < 2; ++mi)
#pragma unroll
        for (int ni = 0; ni < 2; ++ni)
            acc[mi][ni] = (v8f){};

#ifdef HAVE_TDM
    const unsigned lds_base = (unsigned)(size_t)smem;
    if (wv == 0) tdm_load_tile(q  + (size_t)mblk * Hdim, lds_base, AROWS);
    if (wv == 1) tdm_load_tile(wt + (size_t)nblk * Hdim, lds_base + ASZ, BROWS);
#endif

    for (int i = 0; i < Hdim / KSTEP; ++i) {
#ifdef HAVE_TDM
        const int cur = i & 1;
        const int nxt = cur ^ 1;
        if (i + 1 < Hdim / KSTEP) {
            const int k = (i + 1) * KSTEP;
            if (wv == 0) tdm_load_tile(q  + (size_t)mblk * Hdim + k, lds_base + nxt * BUFSZ, AROWS);
            if (wv == 1) tdm_load_tile(wt + (size_t)nblk * Hdim + k, lds_base + nxt * BUFSZ + ASZ, BROWS);
            if (wv <= 1) __builtin_amdgcn_s_wait_tensorcnt(1);
        } else {
            if (wv <= 1) __builtin_amdgcn_s_wait_tensorcnt(0);
        }
        __syncthreads();
        const unsigned char* As = smem + cur * BUFSZ;
        const unsigned char* Bs = smem + cur * BUFSZ + ASZ;
#else
        __syncthreads();
        for (int idx = tid; idx < AROWS * 8; idx += 256) {
            int r = idx >> 3, c = idx & 7;
            *(uint4*)(smem + r * PITCH + c * 16) =
                *(const uint4*)(q + (size_t)(mblk + r) * Hdim + i * KSTEP + c * 16);
        }
        for (int idx = tid; idx < BROWS * 8; idx += 256) {
            int r = idx >> 3, c = idx & 7;
            *(uint4*)(smem + ASZ + r * PITCH + c * 16) =
                *(const uint4*)(wt + (size_t)(nblk + r) * Hdim + i * KSTEP + c * 16);
        }
        __syncthreads();
        const unsigned char* As = smem;
        const unsigned char* Bs = smem + ASZ;
#endif

        v16i af[2], bf[2];
#pragma unroll
        for (int mi = 0; mi < 2; ++mi) {
            const unsigned char* p = As + (unsigned)(wm * 32 + mi * 16 + l15) * PITCH + lhi * 16;
#pragma unroll
            for (int c = 0; c < 8; ++c) {
                unsigned long long d = *(const unsigned long long*)(p + 32 * (c >> 1) + 8 * (c & 1));
                af[mi][2 * c]     = (int)(unsigned)(d & 0xffffffffull);
                af[mi][2 * c + 1] = (int)(unsigned)(d >> 32);
            }
        }
#pragma unroll
        for (int ni = 0; ni < 2; ++ni) {
            const unsigned char* p = Bs + (unsigned)(wn * 32 + ni * 16 + l15) * PITCH + lhi * 16;
#pragma unroll
            for (int c = 0; c < 4; ++c) {
                uint4 e = *(const uint4*)(p + 32 * c);
                bf[ni][4 * c]     = (int)e.x;
                bf[ni][4 * c + 1] = (int)e.y;
                bf[ni][4 * c + 2] = (int)e.z;
                bf[ni][4 * c + 3] = (int)e.w;
            }
        }
#pragma unroll
        for (int mi = 0; mi < 2; ++mi)
#pragma unroll
            for (int ni = 0; ni < 2; ++ni)
                acc[mi][ni] = __builtin_amdgcn_wmma_f32_16x16x128_fp8_fp8(
                    af[mi], bf[ni], (short)0, acc[mi][ni], false, false);

#ifdef HAVE_TDM
        __syncthreads();
#endif
    }

    const float s = scale[l] * wscale[l];
#pragma unroll
    for (int mi = 0; mi < 2; ++mi)
#pragma unroll
        for (int ni = 0; ni < 2; ++ni)
#pragma unroll
            for (int r = 0; r < 8; ++r) {
                const int row = mblk + wm * 32 + mi * 16 + r + 8 * lhi;
                const int col = nblk + wn * 32 + ni * 16 + l15;
                float* p = resid + (size_t)row * Hdim + col;
                *p += acc[mi][ni][r] * s;
            }
}

extern "C" void kernel_launch(void* const* d_in, const int* in_sizes, int n_in,
                              void* d_out, int out_size, void* d_ws, size_t ws_size,
                              hipStream_t stream) {
    const float* x      = (const float*)d_in[0];
    const float* norm_w = (const float*)d_in[1];
    const float* W      = (const float*)d_in[2];
    const float* wscale = (const float*)d_in[3];
    const float* scale  = (const float*)d_in[4];
    float* out = (float*)d_out;

    char* ws = (char*)d_ws;
    float* resid      = (float*)ws;
    unsigned char* qb = (unsigned char*)(ws + (size_t)Tdim * Hdim * 4);
    unsigned char* wq = (unsigned char*)(ws + (size_t)Tdim * Hdim * 4
                                            + (size_t)Tdim * Hdim);

    dim3 rb(256), rg(Tdim);
    dim3 cb(32, 8), cg(Hdim / 32, Hdim / 32);
    dim3 gb(256), gg(Hdim / 64, Tdim / 128);

    relu_rms_quant_kernel<<<rg, rb, 0, stream>>>(x, norm_w, scale, resid, qb);
    for (int l = 0; l < 3; ++l) {
        wconv_kernel<<<cg, cb, 0, stream>>>(W + (size_t)l * Hdim * Hdim, wq);
        gemm_fp8_kernel<<<gg, gb, 0, stream>>>(qb, wq, resid, scale, wscale, l);
        if (l < 2)
            rms_quant_kernel<<<rg, rb, 0, stream>>>(resid,
                                                    norm_w + (size_t)(l + 1) * Hdim,
                                                    scale + (l + 1), qb);
        else
            rms_out_kernel<<<rg, rb, 0, stream>>>(resid,
                                                  norm_w + (size_t)3 * Hdim, out);
    }
}
